// KimiDeltaAttention_14345190768770
// MI455X (gfx1250) — compile-verified, matrix operand order checked by reading, not run on hardware
//
#include <hip/hip_runtime.h>
#include <hip/hip_bf16.h>
#include <math.h>

typedef __attribute__((ext_vector_type(16))) _Float16 v16h;
typedef __attribute__((ext_vector_type(8)))  float    v8f;

#define B_   2
#define T_   2048
#define HID_ 2048
#define H_   16
#define DK_  128
#define DV_  128
#define M_   (B_ * T_)   // 4096 rows

// ---------------------------------------------------------------------------
// Async global->LDS copy (CDNA5 GLOBAL_LOAD_ASYNC_TO_LDS_B32, ASYNCcnt), with
// a synchronous fallback of identical semantics if the builtin is absent.
// Probe feedback: the builtin takes address-space-qualified int* params.
// ---------------------------------------------------------------------------
#if defined(__has_builtin)
#  if __has_builtin(__builtin_amdgcn_global_load_async_to_lds_b32)
#    define KDA_ASYNC_LDS 1
#  endif
#endif

typedef __attribute__((address_space(1))) int as1_int;
typedef __attribute__((address_space(3))) int as3_int;

__device__ __forceinline__ void ld_to_lds_f32(const float* g, float* l) {
#ifdef KDA_ASYNC_LDS
  __builtin_amdgcn_global_load_async_to_lds_b32(
      (as1_int*)(void*)g, (as3_int*)l, 0, 0);
#else
  *l = *g;
#endif
}

__device__ __forceinline__ void lds_copy_wait() {
#ifdef KDA_ASYNC_LDS
#  if __has_builtin(__builtin_amdgcn_s_wait_asynccnt)
  __builtin_amdgcn_s_wait_asynccnt(0);
#  else
  asm volatile("s_wait_asynccnt 0x0" ::: "memory");
#  endif
#endif
}

// ---------------------------------------------------------------------------
// GEMM: Y[M,N] = X[M,K] @ W[N,K]^T   (fp32 in/out, f16 WMMA core, f32 accum)
// Tile 128x128, K-step 32, 256 threads = 8 waves, each wave: 16 rows x 128 cols
// Double-buffered LDS + register-staged next tile.
// ---------------------------------------------------------------------------
#define TM  128
#define TN  128
#define TKK 32

__global__ __launch_bounds__(256) void gemm_xwt_f16(
    const float* __restrict__ X, const float* __restrict__ W,
    float* __restrict__ Y, int Mdim, int Ndim, int Kdim)
{
  __shared__ _Float16 As[2][TM][TKK];   // 2 x 8 KB
  __shared__ _Float16 Bs[2][TN][TKK];   // 2 x 8 KB
  const int tid  = threadIdx.x;
  const int wid  = tid >> 5;
  const int lane = tid & 31;
  const int m0 = blockIdx.y * TM;
  const int n0 = blockIdx.x * TN;

  v8f acc[8];
  #pragma unroll
  for (int c = 0; c < 8; ++c) acc[c] = (v8f){0,0,0,0,0,0,0,0};

  const int lr = tid >> 1;          // tile row 0..127
  const int lc = (tid & 1) * 16;    // 0 or 16 within K-step
  const bool brow_ok = (n0 + lr) < Ndim;

  // ---- prologue: stage tile 0 into buffer 0 ----
  {
    const float* xp = X + (size_t)(m0 + lr) * Kdim + lc;
    #pragma unroll
    for (int j = 0; j < 16; ++j) As[0][lr][lc + j] = (_Float16)xp[j];
    if (brow_ok) {
      const float* wp = W + (size_t)(n0 + lr) * Kdim + lc;
      #pragma unroll
      for (int j = 0; j < 16; ++j) Bs[0][lr][lc + j] = (_Float16)wp[j];
    } else {
      #pragma unroll
      for (int j = 0; j < 16; ++j) Bs[0][lr][lc + j] = (_Float16)0.0f;
    }
  }
  __syncthreads();

  const int nsteps = Kdim / TKK;
  const int lrow = lane & 15;
  const int khi  = (lane >> 4) << 3;   // 0 or 8

  for (int s = 0; s < nsteps; ++s) {
    const int p = s & 1;
    const bool has_next = (s + 1) < nsteps;

    // issue next tile's global loads early (latency hidden by WMMA block)
    float xr[16], wr[16];
    if (has_next) {
      const int ktn = (s + 1) * TKK;
      const float* xp = X + (size_t)(m0 + lr) * Kdim + ktn + lc;
      #pragma unroll
      for (int j = 0; j < 16; ++j) xr[j] = xp[j];
      if (brow_ok) {
        const float* wp = W + (size_t)(n0 + lr) * Kdim + ktn + lc;
        #pragma unroll
        for (int j = 0; j < 16; ++j) wr[j] = wp[j];
      } else {
        #pragma unroll
        for (int j = 0; j < 16; ++j) wr[j] = 0.0f;
      }
      if (s + 2 < nsteps) {  // prefetch tile s+2 -> global_prefetch_b8
        __builtin_prefetch(X + (size_t)(m0 + lr) * Kdim + (s + 2) * TKK + lc, 0, 0);
        if (brow_ok)
          __builtin_prefetch(W + (size_t)(n0 + lr) * Kdim + (s + 2) * TKK + lc, 0, 0);
      }
    }

    // fragment lane mapping per ISA 7.12.2 (16-bit A 16x32)
    v16h a;
    #pragma unroll
    for (int j = 0; j < 16; ++j)
      a[j] = As[p][wid * 16 + lrow][((j < 8) ? j : j + 8) + khi];

    v16h bf[8];
    #pragma unroll
    for (int c = 0; c < 8; ++c) {
      #pragma unroll
      for (int j = 0; j < 16; ++j)
        bf[c][j] = Bs[p][c * 16 + lrow][((j < 8) ? j : j + 8) + khi];
    }
    #pragma unroll
    for (int c = 0; c < 8; ++c)
      acc[c] = __builtin_amdgcn_wmma_f32_16x16x32_f16(
          false, a, false, bf[c], (short)0, acc[c], false, false);

    // convert + store next tile into the other buffer
    if (has_next) {
      const int q = p ^ 1;
      #pragma unroll
      for (int j = 0; j < 16; ++j) As[q][lr][lc + j] = (_Float16)xr[j];
      #pragma unroll
      for (int j = 0; j < 16; ++j) Bs[q][lr][lc + j] = (_Float16)wr[j];
    }
    __syncthreads();
  }

  // D layout: VGPR r -> M = r + 8*(lane>=16), N = n-tile + (lane&15)
  const int mhalf = (lane >> 4) * 8;
  #pragma unroll
  for (int c = 0; c < 8; ++c) {
    const int n = n0 + c * 16 + lrow;
    if (n < Ndim) {
      #pragma unroll
      for (int r = 0; r < 8; ++r) {
        const int m = m0 + wid * 16 + r + mhalf;
        if (m < Mdim) Y[(size_t)m * Ndim + n] = acc[c][r];
      }
    }
  }
}

// ---------------------------------------------------------------------------
__device__ __forceinline__ float siluf(float x) { return x / (1.f + __expf(-x)); }

// causal depthwise conv (KC=4) + SiLU + per-head L2 norm for q and k
__global__ __launch_bounds__(128) void conv_qk_norm(
    const float* __restrict__ qp, const float* __restrict__ kp,
    const float* __restrict__ cwq, const float* __restrict__ cwk,
    float* __restrict__ qn, float* __restrict__ kn)
{
  __shared__ float red[128];
  const int bh = blockIdx.x;        // (b*T+t)*H + h
  const int bt = bh / H_;
  const int h  = bh % H_;
  const int t  = bt % T_;
  const int d  = h * DK_ + threadIdx.x;
  float qv = 0.f, kv = 0.f;
  #pragma unroll
  for (int i = 0; i < 4; ++i) {
    const int tt = t - 3 + i;
    if (tt >= 0) {
      const size_t idx = (size_t)(bt + tt - t) * HID_ + d;  // (b*T+tt)
      qv = fmaf(qp[idx], cwq[d * 4 + i], qv);
      kv = fmaf(kp[idx], cwk[d * 4 + i], kv);
    }
  }
  qv = siluf(qv); kv = siluf(kv);

  red[threadIdx.x] = qv * qv; __syncthreads();
  #pragma unroll
  for (int s = 64; s > 0; s >>= 1) {
    if (threadIdx.x < s) red[threadIdx.x] += red[threadIdx.x + s];
    __syncthreads();
  }
  const float qss = red[0]; __syncthreads();
  red[threadIdx.x] = kv * kv; __syncthreads();
  #pragma unroll
  for (int s = 64; s > 0; s >>= 1) {
    if (threadIdx.x < s) red[threadIdx.x] += red[threadIdx.x + s];
    __syncthreads();
  }
  const float kss = red[0];
  const size_t o = (size_t)bt * HID_ + d;
  qn[o] = qv * rsqrtf(qss + 1e-6f) * 0.08838834764831845f;  // * DK^-0.5
  kn[o] = kv * rsqrtf(kss + 1e-6f);
}

// conv + SiLU for v (pure elementwise over channels)
__global__ __launch_bounds__(256) void conv_v(
    const float* __restrict__ vp, const float* __restrict__ cwv,
    float* __restrict__ vv)
{
  const int idx = blockIdx.x * 256 + threadIdx.x;   // < M_*HID_
  const int d  = idx % HID_;
  const int bt = idx / HID_;
  const int t  = bt % T_;
  float a = 0.f;
  #pragma unroll
  for (int i = 0; i < 4; ++i) {
    const int tt = t - 3 + i;
    if (tt >= 0) a = fmaf(vp[(size_t)(bt + tt - t) * HID_ + d], cwv[d * 4 + i], a);
  }
  vv[idx] = siluf(a);
}

// g = -exp(A_log[h]) * softplus(fb + dt_bias); store exp(g) in-place
__global__ __launch_bounds__(256) void gate_kernel(
    float* __restrict__ fb, const float* __restrict__ dtb,
    const float* __restrict__ A_log)
{
  const int idx = blockIdx.x * 256 + threadIdx.x;   // < M_*HID_
  const int d = idx % HID_;
  const int h = d >> 7;
  const float x = fb[idx] + dtb[d];
  const float sp = (x > 20.f) ? x : log1pf(__expf(x));
  fb[idx] = __expf(-__expf(A_log[h]) * sp);
}

__global__ __launch_bounds__(256) void beta_kernel(float* __restrict__ bp)
{
  const int idx = blockIdx.x * 256 + threadIdx.x;   // < M_*H_
  bp[idx] = 1.f / (1.f + __expf(-bp[idx]));
}

// ---------------------------------------------------------------------------
// Gated delta-rule scan. One block per (b,h); thread v owns S[:,v] (128 VGPRs).
// Per-timestep k/q/exp(g)/v vectors double-buffered in LDS and filled with
// async global->LDS copies for step t+1 while step t computes.
// ---------------------------------------------------------------------------
__global__ __launch_bounds__(128) void kda_scan(
    const float* __restrict__ qn, const float* __restrict__ kn,
    const float* __restrict__ vv, const float* __restrict__ eg,
    const float* __restrict__ beta, float* __restrict__ o)
{
  __shared__ float sk[2][128], sq[2][128], sg[2][128], sv[2][128];
  const int b = blockIdx.x / H_;
  const int h = blockIdx.x % H_;
  const int v = threadIdx.x;
  float S[128];
  #pragma unroll
  for (int i = 0; i < 128; ++i) S[i] = 0.f;

  const size_t base0 = (size_t)(b * T_) * HID_ + h * DK_;

  // preload t = 0 into buffer 0
  ld_to_lds_f32(kn + base0 + v, &sk[0][v]);
  ld_to_lds_f32(qn + base0 + v, &sq[0][v]);
  ld_to_lds_f32(eg + base0 + v, &sg[0][v]);
  ld_to_lds_f32(vv + base0 + v, &sv[0][v]);
  lds_copy_wait();
  __syncthreads();

  for (int t = 0; t < T_; ++t) {
    const int buf = t & 1;
    const size_t base = base0 + (size_t)t * HID_;

    // issue async copies of step t+1 into the other buffer
    if (t + 1 < T_) {
      const size_t nb = base + HID_;
      ld_to_lds_f32(kn + nb + v, &sk[buf ^ 1][v]);
      ld_to_lds_f32(qn + nb + v, &sq[buf ^ 1][v]);
      ld_to_lds_f32(eg + nb + v, &sg[buf ^ 1][v]);
      ld_to_lds_f32(vv + nb + v, &sv[buf ^ 1][v]);
    }
    const float bt = beta[(size_t)(b * T_ + t) * H_ + h];

    float kv = 0.f;
    #pragma unroll
    for (int k2 = 0; k2 < 128; ++k2) {
      const float s = sg[buf][k2] * S[k2];   // decay along key dim
      S[k2] = s;
      kv = fmaf(sk[buf][k2], s, kv);
    }
    const float u = (sv[buf][v] - kv) * bt;
    float ov = 0.f;
    #pragma unroll
    for (int k2 = 0; k2 < 128; ++k2) {
      const float s = fmaf(sk[buf][k2], u, S[k2]);
      S[k2] = s;
      ov = fmaf(sq[buf][k2], s, ov);
    }
    o[base + v] = ov;

    lds_copy_wait();     // t+1 data landed in LDS
    __syncthreads();     // visible to all waves; safe to reuse buffers
  }
}

// RMS norm over DV * weight * sigmoid(g2), in-place on o
__global__ __launch_bounds__(128) void norm_gate(
    float* __restrict__ o, const float* __restrict__ gb,
    const float* __restrict__ onw)
{
  __shared__ float red[128];
  const int bh = blockIdx.x;
  const int bt = bh / H_;
  const int h  = bh % H_;
  const size_t idx = (size_t)bt * HID_ + h * DV_ + threadIdx.x;
  const float ov = o[idx];
  red[threadIdx.x] = ov * ov; __syncthreads();
  #pragma unroll
  for (int s = 64; s > 0; s >>= 1) {
    if (threadIdx.x < s) red[threadIdx.x] += red[threadIdx.x + s];
    __syncthreads();
  }
  const float r = rsqrtf(red[0] * (1.f / 128.f) + 1e-5f);
  o[idx] = ov * r * onw[threadIdx.x] / (1.f + __expf(-gb[idx]));
}

// ---------------------------------------------------------------------------
extern "C" void kernel_launch(void* const* d_in, const int* in_sizes, int n_in,
                              void* d_out, int out_size, void* d_ws, size_t ws_size,
                              hipStream_t stream)
{
  (void)in_sizes; (void)n_in; (void)out_size; (void)ws_size;
  const float* h    = (const float*)d_in[0];
  const float* Wq   = (const float*)d_in[2];
  const float* Wk   = (const float*)d_in[3];
  const float* Wv   = (const float*)d_in[4];
  const float* cwq  = (const float*)d_in[5];
  const float* cwk  = (const float*)d_in[6];
  const float* cwv  = (const float*)d_in[7];
  const float* A_log= (const float*)d_in[8];
  const float* Wfa  = (const float*)d_in[9];
  const float* Wfb  = (const float*)d_in[10];
  const float* dtb  = (const float*)d_in[11];
  const float* Wb   = (const float*)d_in[12];
  const float* Wga  = (const float*)d_in[13];
  const float* Wgb  = (const float*)d_in[14];
  const float* onw  = (const float*)d_in[15];
  const float* Wo   = (const float*)d_in[16];
  float* out = (float*)d_out;

  char* ws = (char*)d_ws;
  const size_t SZ = (size_t)M_ * HID_ * sizeof(float);            // 32 MB
  float* qproj = (float*)(ws + 0 * SZ);
  float* kproj = (float*)(ws + 1 * SZ);
  float* vproj = (float*)(ws + 2 * SZ);
  float* qn    = (float*)(ws + 3 * SZ);
  float* kn    = (float*)(ws + 4 * SZ);
  float* vvb   = (float*)(ws + 5 * SZ);
  float* eg    = (float*)(ws + 6 * SZ);   // fb GEMM out -> exp(g) in-place
  float* gbuf  = (float*)(ws + 7 * SZ);   // gb GEMM out
  float* obuf  = (float*)(ws + 8 * SZ);   // scan out -> gated in-place
  const size_t SZS = (size_t)M_ * DV_ * sizeof(float);            // 2 MB
  float* fa    = (float*)(ws + 9 * SZ);
  float* ga    = (float*)(ws + 9 * SZ + SZS);
  float* betab = (float*)(ws + 9 * SZ + 2 * SZS);                 // M_*H_ floats

  const dim3 blk(256);
  const dim3 g_big(HID_ / TN, M_ / TM);   // 16 x 32
  const dim3 g_n1(1, M_ / TM);            // N <= 128

  // projections
  gemm_xwt_f16<<<g_big, blk, 0, stream>>>(h, Wq, qproj, M_, HID_, HID_);
  gemm_xwt_f16<<<g_big, blk, 0, stream>>>(h, Wk, kproj, M_, HID_, HID_);
  gemm_xwt_f16<<<g_big, blk, 0, stream>>>(h, Wv, vproj, M_, HID_, HID_);
  gemm_xwt_f16<<<g_n1,  blk, 0, stream>>>(h, Wfa, fa,   M_, DV_,  HID_);
  gemm_xwt_f16<<<g_n1,  blk, 0, stream>>>(h, Wga, ga,   M_, DV_,  HID_);
  gemm_xwt_f16<<<g_n1,  blk, 0, stream>>>(h, Wb,  betab,M_, H_,   HID_);
  gemm_xwt_f16<<<g_big, blk, 0, stream>>>(fa, Wfb, eg,   M_, HID_, DV_);
  gemm_xwt_f16<<<g_big, blk, 0, stream>>>(ga, Wgb, gbuf, M_, HID_, DV_);

  // elementwise / conv / gates
  conv_qk_norm<<<M_ * H_, 128, 0, stream>>>(qproj, kproj, cwq, cwk, qn, kn);
  conv_v<<<(M_ * HID_) / 256, 256, 0, stream>>>(vproj, cwv, vvb);
  gate_kernel<<<(M_ * HID_) / 256, 256, 0, stream>>>(eg, dtb, A_log);
  beta_kernel<<<(M_ * H_) / 256, 256, 0, stream>>>(betab);

  // sequential gated delta-rule scan (parallel over B*H and DV columns)
  kda_scan<<<B_ * H_, 128, 0, stream>>>(qn, kn, vvb, eg, betab, obuf);

  // output gating + final projection
  norm_gate<<<M_ * H_, 128, 0, stream>>>(obuf, gbuf, onw);
  gemm_xwt_f16<<<g_big, blk, 0, stream>>>(obuf, Wo, out, M_, HID_, HID_);
}
